// TinyMNISTQAT_2027224563934
// MI455X (gfx1250) — matrix operand order repaired and checked by reading, not compiled, not run on hardware
//
#include <hip/hip_runtime.h>
#include <hip/hip_bf16.h>

typedef __attribute__((ext_vector_type(16))) _Float16 v16h;
typedef __attribute__((ext_vector_type(8)))  _Float16 v8h;
typedef __attribute__((ext_vector_type(8)))  float    v8f;

#define QMAXF 32767.0f
#define HW 224
#define NPIX (HW*HW)                 // 50176
#define BATCH 64
#define TILES_W 14                   // 224/16
#define TILES_PER_IMG (HW*TILES_W)   // 3136
#define NTILES (BATCH*TILES_PER_IMG) // 200704
#define TPW 4                        // tiles per wave

// ---------------- device helpers ----------------

// fake-quant with a divide (low-volume prep/fc paths only)
__device__ __forceinline__ float fq(float x, float s) {
    float r = rintf(x / s);
    r = fminf(fmaxf(r, -QMAXF), QMAXF);
    return r * s;
}

// divide-free h-GELU constants; scale = 2^clip(round(shift),4,12)
struct HG {
    float scale, three, six, inv_six, inv_scale;
};
__device__ __forceinline__ HG make_hg(float shift_param) {
    float s = fminf(fmaxf(rintf(shift_param), 4.0f), 12.0f);
    int si = (int)s;
    HG g;
    g.scale     = __int_as_float((127 + si) << 23);   // 2^si
    g.inv_scale = __int_as_float((127 - si) << 23);   // exact 2^-si
    g.three     = 3.0f * g.scale;
    g.six       = 6.0f * g.scale;
    g.inv_six   = 0.16666667f * g.inv_scale;          // (1/6)*2^-si, exact rescale
    return g;
}

// h-GELU core, returns the clipped integer yi (output value is yi * 2^-si)
__device__ __forceinline__ float hgelu_int(float x, const HG& g) {
    float xi    = fminf(fmaxf(rintf(x * g.scale), -32768.0f), 32767.0f);
    float slope = rintf(xi * 1.703125f);              // 218/128
    float gate  = fminf(fmaxf(slope + g.three, 0.0f), g.six);
    float yi    = rintf(xi * gate * g.inv_six);
    return fminf(fmaxf(yi, -QMAXF), QMAXF);
}

// Lane's A fragment: pre-swizzled so it is one contiguous 32B chunk.
__device__ __forceinline__ v16h load_a(const _Float16* __restrict__ Ap, int lane) {
    return *(const v16h*)(Ap + lane * 16);
}

// One pixel's 16 channels (NHWC f16, C=16 -> 32B contiguous), zero outside image.
__device__ __forceinline__ v16h load_b_pix(const _Float16* __restrict__ act,
                                           int n, int hy, int wx) {
    if ((unsigned)hy < (unsigned)HW && (unsigned)wx < (unsigned)HW)
        return *(const v16h*)(act + (((size_t)n * HW + hy) * HW + wx) * 16);
    v16h z = {};
    return z;
}

// ---------------- prep: weight fake-quant + A-layout swizzle ----------------
// A (16-bit, 16x32) per-lane layout: lane L -> M = L&15, h8 = L>>4,
// v16h elements 0..15 = K = h8*16 + 0..15
// K meaning: tap = 2p + K/16, input channel c = K%16 (conv1: K = tap, c = 0).
__global__ void prep_kernel(const float* __restrict__ w1, const float* __restrict__ w2,
                            const float* __restrict__ w3,
                            _Float16* __restrict__ A1, _Float16* __restrict__ A2,
                            _Float16* __restrict__ A3) {
    __shared__ float red[256];
    __shared__ float wsc[3];
    const int tid = threadIdx.x;
    const float* wp[3] = {w1, w2, w3};
    const int   wn[3] = {16 * 9, 16 * 16 * 9, 16 * 16 * 9};
    for (int l = 0; l < 3; ++l) {
        float m = 0.0f;
        for (int i = tid; i < wn[l]; i += 256) m = fmaxf(m, fabsf(wp[l][i]));
        red[tid] = m; __syncthreads();
        for (int s = 128; s > 0; s >>= 1) {
            if (tid < s) red[tid] = fmaxf(red[tid], red[tid + s]);
            __syncthreads();
        }
        if (tid == 0) wsc[l] = fmaxf(red[0] / QMAXF, 1e-8f);
        __syncthreads();
    }
    for (int idx = tid; idx < 32 * 16; idx += 256) {
        int lane = idx >> 4, e = idx & 15;
        int m = lane & 15, h8 = lane >> 4;
        int K = h8 * 16 + e;
        float v = (K < 9) ? fq(w1[m * 9 + K], wsc[0]) : 0.0f;
        A1[idx] = (_Float16)v;
    }
    for (int idx = tid; idx < 5 * 32 * 16; idx += 256) {
        int p = idx >> 9, lane = (idx >> 4) & 31, e = idx & 15;
        int m = lane & 15, h8 = lane >> 4;
        int K = h8 * 16 + e;
        int tap = 2 * p + (K >> 4);
        int c = K & 15;
        float v2 = 0.0f, v3 = 0.0f;
        if (tap < 9) {
            int wi = ((m * 16 + c) * 3 + tap / 3) * 3 + (tap % 3);
            v2 = fq(w2[wi], wsc[1]);
            v3 = fq(w3[wi], wsc[2]);
        }
        A2[idx] = (_Float16)v2;
        A3[idx] = (_Float16)v3;
    }
}

// ---------------- conv1 (C_in = 1, K = 9 taps in one WMMA), 4 tiles/wave ----------------
__global__ __launch_bounds__(256) void conv1_kernel(
    const float* __restrict__ x, const _Float16* __restrict__ A1,
    const float* __restrict__ bias, const float* __restrict__ a_scales,
    const float* __restrict__ shifts, _Float16* __restrict__ actout) {
    const int wave = threadIdx.x >> 5;
    const int lane = threadIdx.x & 31;
    const int tile0 = (blockIdx.x * 8 + wave) * TPW;
    const int nIdx = lane & 15;
    const int half = lane >> 4;

    const float a0     = a_scales[0];
    const float inv_a0 = 1.0f / a0;              // one divide per wave-thread
    const float an     = a_scales[1];
    const HG    g      = make_hg(shifts[0]);
    const float cfold  = g.inv_scale / an;       // yi*2^-si -> units of a_next
    const v16h  a      = load_a(A1, lane);
    const v8f   bv     = *(const v8f*)(bias + (half << 3));

    for (int t = 0; t < TPW; ++t) {
        const int tile = tile0 + t;
        const int n  = tile / TILES_PER_IMG;
        const int rr = tile % TILES_PER_IMG;
        const int h  = rr / TILES_W;
        const int w0 = (rr % TILES_W) * 16;

        // B: lanes 0-15 hold K=0..15 (taps 0..8 of pixel nIdx); lanes 16-31 -> zeros
        v16h b = {};
        if (half == 0) {
            const int w = w0 + nIdx;
            #pragma unroll
            for (int tap = 0; tap < 9; ++tap) {
                int hy = h + tap / 3 - 1;
                int wx = w + tap % 3 - 1;
                float v = 0.0f;
                if ((unsigned)hy < (unsigned)HW && (unsigned)wx < (unsigned)HW) {
                    float r = rintf(x[((size_t)n * HW + hy) * HW + wx] * inv_a0);
                    v = fminf(fmaxf(r, -QMAXF), QMAXF) * a0;
                }
                b[tap] = (_Float16)v;
            }
        }

        v8f acc = {};
        acc = __builtin_amdgcn_wmma_f32_16x16x32_f16(false, a, false, b,
                                                     (short)0, acc, false, false);

        const size_t pix = ((size_t)n * HW + h) * HW + (w0 + nIdx);
        v8h o;
        #pragma unroll
        for (int r = 0; r < 8; ++r) {
            float yi = hgelu_int(acc[r] + bv[r], g);
            float q  = fminf(fmaxf(rintf(yi * cfold), -QMAXF), QMAXF);
            o[r] = (_Float16)(q * an);                   // folded next-layer fq
        }
        *(v8h*)(actout + pix * 16 + (half << 3)) = o;
    }
}

// ---------------- conv2 (16->16, 5 WMMAs/tile, store), 4 tiles/wave ----------------
__global__ __launch_bounds__(256) void conv_mid_kernel(
    const _Float16* __restrict__ actin, const _Float16* __restrict__ Ap,
    const float* __restrict__ bias, const float* __restrict__ a_scales,
    const float* __restrict__ shifts, _Float16* __restrict__ actout) {
    const int wave = threadIdx.x >> 5;
    const int lane = threadIdx.x & 31;
    const int tile0 = (blockIdx.x * 8 + wave) * TPW;
    const int nIdx = lane & 15;
    const int half = lane >> 4;

    const float an    = a_scales[2];
    const HG    g     = make_hg(shifts[1]);
    const float cfold = g.inv_scale / an;
    const v8f   bv    = *(const v8f*)(bias + (half << 3));

    // A fragments held in registers across all 4 tiles (5 x v16h = 40 VGPRs)
    v16h a0 = load_a(Ap + 0 * 512, lane);
    v16h a1 = load_a(Ap + 1 * 512, lane);
    v16h a2 = load_a(Ap + 2 * 512, lane);
    v16h a3 = load_a(Ap + 3 * 512, lane);
    v16h a4 = load_a(Ap + 4 * 512, lane);

    for (int t = 0; t < TPW; ++t) {
        const int tile = tile0 + t;
        const int n  = tile / TILES_PER_IMG;
        const int rr = tile % TILES_PER_IMG;
        const int h  = rr / TILES_W;
        const int w  = (rr % TILES_W) * 16 + nIdx;

        v8f acc = {};
        // tap for pair p = 2p + half; p=4,half=1 -> zero block
        v16h b;
        b = load_b_pix(actin, n, h + (0 + half) / 3 - 1, w + (0 + half) % 3 - 1);
        acc = __builtin_amdgcn_wmma_f32_16x16x32_f16(false, a0, false, b, (short)0, acc, false, false);
        b = load_b_pix(actin, n, h + (2 + half) / 3 - 1, w + (2 + half) % 3 - 1);
        acc = __builtin_amdgcn_wmma_f32_16x16x32_f16(false, a1, false, b, (short)0, acc, false, false);
        b = load_b_pix(actin, n, h + (4 + half) / 3 - 1, w + (4 + half) % 3 - 1);
        acc = __builtin_amdgcn_wmma_f32_16x16x32_f16(false, a2, false, b, (short)0, acc, false, false);
        b = load_b_pix(actin, n, h + (6 + half) / 3 - 1, w + (6 + half) % 3 - 1);
        acc = __builtin_amdgcn_wmma_f32_16x16x32_f16(false, a3, false, b, (short)0, acc, false, false);
        { v16h bz = {};
          if (half == 0) bz = load_b_pix(actin, n, h + 8 / 3 - 1, w + 8 % 3 - 1);
          acc = __builtin_amdgcn_wmma_f32_16x16x32_f16(false, a4, false, bz, (short)0, acc, false, false); }

        const size_t pix = ((size_t)n * HW + h) * HW + w;
        v8h o;
        #pragma unroll
        for (int r = 0; r < 8; ++r) {
            float yi = hgelu_int(acc[r] + bv[r], g);
            float q  = fminf(fmaxf(rintf(yi * cfold), -QMAXF), QMAXF);
            o[r] = (_Float16)(q * an);
        }
        *(v8h*)(actout + pix * 16 + (half << 3)) = o;
    }
}

// ---------------- conv3 + per-tile pool partials (no activation store) ----------------
__global__ __launch_bounds__(256) void conv_pool_kernel(
    const _Float16* __restrict__ actin, const _Float16* __restrict__ Ap,
    const float* __restrict__ bias, const float* __restrict__ shifts,
    float* __restrict__ part) {
    const int wave = threadIdx.x >> 5;
    const int lane = threadIdx.x & 31;
    const int tile0 = (blockIdx.x * 8 + wave) * TPW;
    const int nIdx = lane & 15;
    const int half = lane >> 4;

    const HG  g  = make_hg(shifts[2]);
    const v8f bv = *(const v8f*)(bias + (half << 3));

    v16h a0 = load_a(Ap + 0 * 512, lane);
    v16h a1 = load_a(Ap + 1 * 512, lane);
    v16h a2 = load_a(Ap + 2 * 512, lane);
    v16h a3 = load_a(Ap + 3 * 512, lane);
    v16h a4 = load_a(Ap + 4 * 512, lane);

    for (int t = 0; t < TPW; ++t) {
        const int tile = tile0 + t;
        const int n  = tile / TILES_PER_IMG;
        const int rr = tile % TILES_PER_IMG;
        const int h  = rr / TILES_W;
        const int w  = (rr % TILES_W) * 16 + nIdx;

        v8f acc = {};
        v16h b;
        b = load_b_pix(actin, n, h + (0 + half) / 3 - 1, w + (0 + half) % 3 - 1);
        acc = __builtin_amdgcn_wmma_f32_16x16x32_f16(false, a0, false, b, (short)0, acc, false, false);
        b = load_b_pix(actin, n, h + (2 + half) / 3 - 1, w + (2 + half) % 3 - 1);
        acc = __builtin_amdgcn_wmma_f32_16x16x32_f16(false, a1, false, b, (short)0, acc, false, false);
        b = load_b_pix(actin, n, h + (4 + half) / 3 - 1, w + (4 + half) % 3 - 1);
        acc = __builtin_amdgcn_wmma_f32_16x16x32_f16(false, a2, false, b, (short)0, acc, false, false);
        b = load_b_pix(actin, n, h + (6 + half) / 3 - 1, w + (6 + half) % 3 - 1);
        acc = __builtin_amdgcn_wmma_f32_16x16x32_f16(false, a3, false, b, (short)0, acc, false, false);
        { v16h bz = {};
          if (half == 0) bz = load_b_pix(actin, n, h + 8 / 3 - 1, w + 8 % 3 - 1);
          acc = __builtin_amdgcn_wmma_f32_16x16x32_f16(false, a4, false, bz, (short)0, acc, false, false); }

        // bias + hgelu, reduce 16 pixels per channel within each half-wave
        #pragma unroll
        for (int r = 0; r < 8; ++r) {
            float v = hgelu_int(acc[r] + bv[r], g) * g.inv_scale;
            #pragma unroll
            for (int mask = 1; mask < 16; mask <<= 1)
                v += __shfl_xor(v, mask, 32);
            if (nIdx == 0) part[(size_t)tile * 16 + (r + (half << 3))] = v;
        }
    }
}

// ---------------- deterministic pool reduce ----------------
__global__ void pool_reduce_kernel(const float* __restrict__ part, float* __restrict__ pool) {
    int idx = blockIdx.x * blockDim.x + threadIdx.x;     // 1024 = 64*16
    if (idx >= BATCH * 16) return;
    int nn = idx >> 4, c = idx & 15;
    const float* p = part + (size_t)nn * TILES_PER_IMG * 16 + c;
    float s = 0.0f;
    for (int t = 0; t < TILES_PER_IMG; ++t) s += p[(size_t)t * 16];
    pool[idx] = s;
}

// ---------------- FC (tiny): mean, fq, GEMV ----------------
__global__ void fc_kernel(const float* __restrict__ pool, const float* __restrict__ fc_w,
                          const float* __restrict__ fc_b, const float* __restrict__ a_scales,
                          float* __restrict__ out) {
    __shared__ float red[256];
    __shared__ float wscale;
    const int tid = threadIdx.x;
    float mx = 0.0f;
    for (int i = tid; i < 160; i += 256) mx = fmaxf(mx, fabsf(fc_w[i]));
    red[tid] = mx; __syncthreads();
    for (int s = 128; s > 0; s >>= 1) {
        if (tid < s) red[tid] = fmaxf(red[tid], red[tid + s]);
        __syncthreads();
    }
    if (tid == 0) wscale = fmaxf(red[0] / QMAXF, 1e-8f);
    __syncthreads();
    const float a3 = a_scales[3];
    for (int idx = tid; idx < BATCH * 10; idx += 256) {
        int nn = idx / 10, o = idx % 10;
        float s = fc_b[o];
        #pragma unroll
        for (int c = 0; c < 16; ++c) {
            float xv = fq(pool[nn * 16 + c] * (1.0f / NPIX), a3);
            float wv = fq(fc_w[o * 16 + c], wscale);
            s += xv * wv;
        }
        out[idx] = s;
    }
}

// ---------------- launch ----------------
extern "C" void kernel_launch(void* const* d_in, const int* in_sizes, int n_in,
                              void* d_out, int out_size, void* d_ws, size_t ws_size,
                              hipStream_t stream) {
    (void)in_sizes; (void)n_in; (void)out_size; (void)ws_size;
    const float* x   = (const float*)d_in[0];
    const float* w1  = (const float*)d_in[1];
    const float* b1  = (const float*)d_in[2];
    const float* w2  = (const float*)d_in[3];
    const float* b2  = (const float*)d_in[4];
    const float* w3  = (const float*)d_in[5];
    const float* b3  = (const float*)d_in[6];
    const float* fcw = (const float*)d_in[7];
    const float* fcb = (const float*)d_in[8];
    const float* asc = (const float*)d_in[9];
    const float* shf = (const float*)d_in[10];
    float* out = (float*)d_out;

    char* ws = (char*)d_ws;
    const size_t actBytes = (size_t)BATCH * NPIX * 16 * sizeof(_Float16); // 102,760,448 (256-aligned)
    _Float16* act0 = (_Float16*)ws;
    _Float16* act1 = (_Float16*)(ws + actBytes);
    char* tail = ws + 2 * actBytes;
    _Float16* A1 = (_Float16*)tail;             // 512 halves
    _Float16* A2 = A1 + 512;                    // 2560 halves
    _Float16* A3 = A2 + 2560;                   // 2560 halves -> tail + 11264 B (256-aligned)
    float* part  = (float*)(tail + 11264);      // NTILES*16 f32 = 12,845,056 B
    float* pool  = (float*)((char*)part + (size_t)NTILES * 16 * sizeof(float));

    prep_kernel<<<1, 256, 0, stream>>>(w1, w2, w3, A1, A2, A3);

    dim3 grid(NTILES / (8 * TPW));   // 6272 blocks, 8 waves x 4 tiles each
    conv1_kernel    <<<grid, 256, 0, stream>>>(x,    A1, b1, asc, shf, act0);
    conv_mid_kernel <<<grid, 256, 0, stream>>>(act0, A2, b2, asc, shf, act1);
    conv_pool_kernel<<<grid, 256, 0, stream>>>(act1, A3, b3, shf, part);
    pool_reduce_kernel<<<4, 256, 0, stream>>>(part, pool);
    fc_kernel<<<1, 256, 0, stream>>>(pool, fcw, fcb, asc, out);
}
